// Interval_Refine_75788992905477
// MI455X (gfx1250) — matrix operand order repaired and checked by reading, not compiled, not run on hardware
//
#include <hip/hip_runtime.h>

#define N_NODES 512
#define DF      128
#define HF      128
#define G3      384     // 3*HF
#define NPS     64
#define NA      192
#define KS      5
#define NCLS    5
#define DBINS   21
#define FOUT    47
#define FIN     259
#define FINP    272     // FIN padded to multiple of 16

typedef __attribute__((ext_vector_type(2)))  float        v2f;
typedef __attribute__((ext_vector_type(8)))  float        v8f;
typedef __attribute__((ext_vector_type(16))) __bf16       v16bf;
typedef __attribute__((ext_vector_type(4)))  unsigned int v4u;
typedef __attribute__((ext_vector_type(8)))  unsigned int v8u;

__device__ __forceinline__ float sigmoidf_(float x) { return 1.0f / (1.0f + expf(-x)); }
__device__ __forceinline__ v8f bcast8(float v) { v8f r = {v, v, v, v, v, v, v, v}; return r; }

// ---------------------------------------------------------------------------
// Kernel 1: x = emb + 0.05*sin(tp*freq); conv5+softmax -> abnormal; tpos
// ---------------------------------------------------------------------------
__global__ void k_prep(const float* __restrict__ emb, const float* __restrict__ tp,
                       const float* __restrict__ npred, const float* __restrict__ kern,
                       const float* __restrict__ audio,
                       float* __restrict__ x, float* __restrict__ abn, float* __restrict__ tpos) {
  int i = blockIdx.x;
  int t = threadIdx.x;
  float tpi = tp[i];
  float freq = 10.0f * (float)t / (float)(DF - 1);
  x[i * DF + t] = emb[i * DF + t] + 0.05f * sinf(tpi * freq);
  if (t == 0) {
    tpos[i] = tpi * audio[0];
    float s[NCLS];
    for (int c = 0; c < NCLS; c++) {
      float acc = 0.f;
      for (int k = 0; k < KS; k++) {
        int j = i + k - KS / 2;
        float v = (j >= 0 && j < N_NODES) ? npred[j * NCLS + c] : 0.f;
        acc += kern[k] * v;
      }
      s[c] = acc;
    }
    float mx = s[0];
    for (int c = 1; c < NCLS; c++) mx = fmaxf(mx, s[c]);
    float se = 0.f;
    for (int c = 0; c < NCLS; c++) se += expf(s[c] - mx);
    abn[i] = expf(s[0] - mx) / se;
  }
}

// ---------------------------------------------------------------------------
// Kernel 2: per-anchor in-interval node list (ordered compaction) + lens.
// One wave32 per anchor; ballot-based stable compaction.
// ---------------------------------------------------------------------------
__global__ void k_order(const float* __restrict__ tpos, const float* __restrict__ anch,
                        int* __restrict__ order, int* __restrict__ lens) {
  int a = blockIdx.x;
  int lane = threadIdx.x;
  float s = anch[a * 2 + 0], e = anch[a * 2 + 1];
  int cnt = 0;
  for (int base = 0; base < N_NODES; base += 32) {
    int i = base + lane;
    float tv = tpos[i];
    bool in = (tv >= s) && (tv <= e);
    unsigned m = (unsigned)__ballot(in);
    if (in) {
      int pos = cnt + __popc(m & ((1u << lane) - 1u));
      order[a * N_NODES + pos] = i;
    }
    cnt += __popc(m);
  }
  if (lane == 0) lens[a] = cnt;
}

// ---------------------------------------------------------------------------
// Kernel 3: GI[sd] = X @ Wih[sd]^T + bih  (fp32 WMMA 16x16x4)
// grid = (32 M-tiles, 6 sd * 24 N-tiles), one wave per 16x16 output tile.
// ---------------------------------------------------------------------------
__global__ void k_gi(const float* __restrict__ x, const float* __restrict__ Wih,
                     const float* __restrict__ bih, float* __restrict__ GI) {
  int mt = blockIdx.x;        // 0..31
  int by = blockIdx.y;        // 0..143
  int sd = by / 24;
  int nt = by % 24;
  int lane = threadIdx.x;
  int half = lane >> 4;
  int l = lane & 15;
  const float* W = Wih + (size_t)sd * G3 * DF;   // [384][128] row-major
  const float* b = bih + sd * G3;
  float* out = GI + (size_t)sd * N_NODES * G3;
  int row = mt * 16 + l;      // A-matrix row for this lane
  int n = nt * 16 + l;        // B/D column for this lane
  v8f acc = bcast8(b[n]);
  for (int kk = 0; kk < DF / 4; kk++) {
    int kb = kk * 4 + half * 2;   // lanes 0-15: K {4k,4k+1}; lanes 16-31: {4k+2,4k+3}
    v2f a;  a.x = x[row * DF + kb];  a.y = x[row * DF + kb + 1];
    v2f bb; bb.x = W[n * DF + kb];   bb.y = W[n * DF + kb + 1];   // B[k][n]=Wih[n][k]
    acc = __builtin_amdgcn_wmma_f32_16x16x4_f32(false, a, false, bb, (short)0, acc,
                                                false, false);
  }
  for (int r = 0; r < 8; r++) {
    int M = mt * 16 + r + half * 8;
    out[(size_t)M * G3 + nt * 16 + l] = acc[r];
  }
}

// ---------------------------------------------------------------------------
// Kernel 3b: convert all 6 Whh matrices f32 -> bf16 in global (TDM source).
// ---------------------------------------------------------------------------
__global__ void k_cvt(const float* __restrict__ Whh_all, __bf16* __restrict__ whh_bf) {
  int idx = blockIdx.x * blockDim.x + threadIdx.x;
  if (idx < 6 * G3 * DF) whh_bf[idx] = (__bf16)Whh_all[idx];
}

// ---------------------------------------------------------------------------
// Kernel 4: batched GRU recurrence. One WG = 16 anchors, one (scale,dir).
// Whh (bf16, 98KB) DMA'd into LDS by the Tensor Data Mover; per step:
// GH = H @ Whh^T via v_wmma_f32_16x16x32_bf16 (24 tiles / 8 waves), then
// elementwise GRU update. Dynamic LDS: whh(98304B)|H(8192B)|GH(24576B).
// ---------------------------------------------------------------------------
__global__ void k_gru_feat(const float* __restrict__ GI, const __bf16* __restrict__ whh_bf_g,
                           const float* __restrict__ bhh_all, const int* __restrict__ order,
                           const int* __restrict__ lens, float* __restrict__ hfeat) {
  extern __shared__ char smem[];
  __bf16* whh = (__bf16*)smem;                               // [384][128] bf16
  float* Hls = (float*)(smem + (size_t)G3 * HF * 2);         // [16][128]
  float* GHls = Hls + 16 * HF;                               // [16][384]
  __shared__ int lensh[16];
  __shared__ int maxlen_s;

  int b = blockIdx.x;          // 0..23
  int s = b >> 3;
  int rem = b & 7;
  int d = rem >> 2;
  int grp = rem & 3;
  int sd = s * 2 + d;
  int abase = s * NPS + grp * 16;
  int tid = threadIdx.x;
  int lane = tid & 31;
  int wave = tid >> 5;         // 0..7
  int half = lane >> 4;
  int l = lane & 15;

  const float* bhh = bhh_all + sd * G3;
  const float* gi_base = GI + (size_t)sd * N_NODES * G3;

  if (tid < 16) lensh[tid] = lens[abase + tid];
  __syncthreads();
  if (tid == 0) {
    int m = 0;
    for (int j = 0; j < 16; j++) m = max(m, lensh[j]);
    maxlen_s = m;
  }
  // --- TDM: DMA Whh[sd] (bf16, 98304B) global -> LDS, issued by wave 0 ----
  if (wave == 0) {
    unsigned long long gaddr =
        (unsigned long long)(const void*)(whh_bf_g + (size_t)sd * G3 * DF);
    unsigned lds_off = (unsigned)(unsigned long long)(void*)whh;
    const unsigned NELEM = (unsigned)(G3 * DF);   // 49152 bf16 elements, one row
    v4u g0;
    g0[0] = 1u;                                   // count=1 (valid), user mode
    g0[1] = lds_off;                              // lds_addr
    g0[2] = (unsigned)(gaddr & 0xFFFFFFFFull);    // global_addr[31:0]
    g0[3] = (unsigned)((gaddr >> 32) & 0x1FFFFFFull) | (2u << 30);  // [56:32]|type=2
    v8u g1;
    g1[0] = (1u << 16);            // workgroup_mask=0, data_size=1 (2 bytes)
    g1[1] = (NELEM & 0xFFFFu) << 16;              // tensor_dim0[15:0]
    g1[2] = (NELEM >> 16) | (1u << 16);           // tensor_dim0[31:16] | tensor_dim1=1
    g1[3] = (NELEM & 0xFFFFu) << 16;              // tile_dim0 = 49152
    g1[4] = 1u;                                   // tile_dim1=1, tile_dim2=0
    g1[5] = NELEM;                                // tensor_dim0_stride[31:0]
    g1[6] = 0u;                                   // stride hi / dim1_stride lo
    g1[7] = 0u;
    v4u gz = {0u, 0u, 0u, 0u};                    // groups 2/3 unused (2D tensor)
    asm volatile("tensor_load_to_lds %0, %1, %2, %3"
                 :: "s"(g0), "s"(g1), "s"(gz), "s"(gz)
                 : "memory");
    __builtin_amdgcn_s_wait_tensorcnt(0);
  }
  for (int idx = tid; idx < 16 * HF; idx += blockDim.x) Hls[idx] = 0.f;
  __syncthreads();
  int maxlen = maxlen_s;

  for (int t = 0; t < maxlen; t++) {
    // A fragments (H, fp32 -> bf16). Lane m=l: half0 gets K {c*32+0..15},
    // half1 gets K {c*32+16..31}.
    v16bf afrag[4];
    for (int c = 0; c < 4; c++) {
      const float* h0 = Hls + l * HF + c * 32 + half * 16;
      const float* h1 = Hls + l * HF + c * 32 + half * 16 + 8;
      for (int i = 0; i < 8; i++) {
        afrag[c][i] = (__bf16)h0[i];
        afrag[c][8 + i] = (__bf16)h1[i];
      }
    }
    // GH = H @ Whh^T + bhh  (each wave: 3 N-tiles x 4 K-chunks of WMMA)
    for (int q = 0; q < 3; q++) {
      int nt = wave * 3 + q;
      int g = nt * 16 + l;
      v8f acc = bcast8(bhh[g]);
      for (int c = 0; c < 4; c++) {
        // B[k][n]=Whh[g][k]; half0: k=c*32+0..15, half1: k=c*32+16..31 (32B aligned)
        const __bf16* wrow = whh + (size_t)g * HF + c * 32 + half * 16;
        v16bf bfrag = *(const v16bf*)wrow;
        acc = __builtin_amdgcn_wmma_f32_16x16x32_bf16(false, afrag[c], false, bfrag,
                                                      (short)0, acc, false, false);
      }
      for (int r = 0; r < 8; r++) GHls[(r + half * 8) * G3 + g] = acc[r];
    }
    __syncthreads();
    // Elementwise GRU update: thread -> (sequence j, 8 hidden units)
    {
      int j = tid >> 4;
      int hb = (tid & 15) * 8;
      int len = lensh[j];
      if (t < len) {
        int a = abase + j;
        int node = order[a * N_NODES + (d == 0 ? t : (len - 1 - t))];
        const float* gi = gi_base + (size_t)node * G3;
        for (int i = 0; i < 8; i++) {
          int h = hb + i;
          float r_ = sigmoidf_(gi[h] + GHls[j * G3 + h]);
          float z_ = sigmoidf_(gi[HF + h] + GHls[j * G3 + HF + h]);
          float n_ = tanhf(gi[2 * HF + h] + r_ * GHls[j * G3 + 2 * HF + h]);
          float ho = Hls[j * HF + h];
          Hls[j * HF + h] = (1.f - z_) * n_ + z_ * ho;
        }
      }
    }
    __syncthreads();
  }
  for (int idx = tid; idx < 16 * HF; idx += blockDim.x) {
    int j = idx / HF, h = idx % HF;
    hfeat[(abase + j) * (2 * HF) + d * HF + h] = Hls[idx];
  }
}

// ---------------------------------------------------------------------------
// Kernel 5: hidden-size-1 abnormal GRU, one thread per (anchor, dir).
// ---------------------------------------------------------------------------
__global__ void k_gru_abn(const float* __restrict__ abn, const float* __restrict__ aWih,
                          const float* __restrict__ aWhh, const float* __restrict__ abih,
                          const float* __restrict__ abhh, const int* __restrict__ order,
                          const int* __restrict__ lens, float* __restrict__ habn) {
  int tid = blockIdx.x * blockDim.x + threadIdx.x;
  if (tid >= NA * 2) return;
  int a = tid >> 1, d = tid & 1;
  int s = a / NPS, sd = s * 2 + d;
  const float* Wi = aWih + sd * 3;
  const float* Wh = aWhh + sd * 3;
  const float* bi = abih + sd * 3;
  const float* bh = abhh + sd * 3;
  int len = lens[a];
  float h = 0.f;
  for (int t = 0; t < len; t++) {
    int node = order[a * N_NODES + (d == 0 ? t : (len - 1 - t))];
    float xv = abn[node];
    float r_ = sigmoidf_(xv * Wi[0] + bi[0] + h * Wh[0] + bh[0]);
    float z_ = sigmoidf_(xv * Wi[1] + bi[1] + h * Wh[1] + bh[1]);
    float n_ = tanhf(xv * Wi[2] + bi[2] + r_ * (h * Wh[2] + bh[2]));
    h = (1.f - z_) * n_ + z_ * h;
  }
  habn[a * 2 + d] = h;
}

// ---------------------------------------------------------------------------
// Kernel 6: MLP head (3 f32-WMMA GEMMs) + softmax offset decode + outputs.
// One WG = 16 anchors (single scale), 8 waves.
// ---------------------------------------------------------------------------
__global__ void k_mlp(const float* __restrict__ hfeat, const float* __restrict__ habn,
                      const float* __restrict__ anch, const float* __restrict__ audio,
                      const float* __restrict__ W1a, const float* __restrict__ b1a,
                      const float* __restrict__ W2a, const float* __restrict__ b2a,
                      const float* __restrict__ W3a, const float* __restrict__ b3a,
                      const float* __restrict__ startw, const float* __restrict__ endw,
                      float* __restrict__ outp) {
  __shared__ float sfb[16 * FINP];
  __shared__ float h1b[16 * 256];
  __shared__ float h2b[16 * 256];
  __shared__ float ob[16 * 48];
  int blk = blockIdx.x;
  int abase = blk * 16;
  int s = abase / NPS;
  int tid = threadIdx.x;
  int lane = tid & 31;
  int wave = tid >> 5;      // 0..7
  int half = lane >> 4;
  int l = lane & 15;
  float al = audio[0];
  const float* W1 = W1a + (size_t)s * FIN * 256;
  const float* b1 = b1a + s * 256;
  const float* W2 = W2a + (size_t)s * 256 * 256;
  const float* b2 = b2a + s * 256;
  const float* W3 = W3a + (size_t)s * 256 * FOUT;
  const float* b3 = b3a + s * FOUT;

  // Build padded feature rows sf = [hf | hb | abn | c/al | w/al | 0-pad]
  for (int idx = tid; idx < 16 * FINP; idx += blockDim.x) {
    int j = idx / FINP, c = idx % FINP;
    int a = abase + j;
    float v = 0.f;
    if (c < 256) v = hfeat[a * 256 + c];
    else if (c == 256) v = 0.5f * (habn[a * 2] + habn[a * 2 + 1]);
    else if (c == 257) v = 0.5f * (anch[a * 2] + anch[a * 2 + 1]) / al;
    else if (c == 258) v = (anch[a * 2 + 1] - anch[a * 2]) / al;
    sfb[j * FINP + c] = v;
  }
  __syncthreads();
  __builtin_prefetch(W2 + tid * 64, 0, 1);  // stream next layer (global_prefetch_b8)

  // Layer 1: [16,272(259)] @ [259,256]
  for (int q = 0; q < 2; q++) {
    int n = (wave * 2 + q) * 16 + l;
    v8f acc = bcast8(b1[n]);
    for (int kk = 0; kk < FINP / 4; kk++) {
      int kb = kk * 4 + half * 2;
      v2f a;  a.x = sfb[l * FINP + kb]; a.y = sfb[l * FINP + kb + 1];
      v2f bb;
      bb.x = (kb < FIN) ? W1[(size_t)kb * 256 + n] : 0.f;
      bb.y = (kb + 1 < FIN) ? W1[(size_t)(kb + 1) * 256 + n] : 0.f;
      acc = __builtin_amdgcn_wmma_f32_16x16x4_f32(false, a, false, bb, (short)0, acc,
                                                  false, false);
    }
    for (int r = 0; r < 8; r++) h1b[(r + half * 8) * 256 + n] = fmaxf(acc[r], 0.f);
  }
  __syncthreads();
  __builtin_prefetch(W3 + tid * 64, 0, 1);

  // Layer 2: [16,256] @ [256,256]
  for (int q = 0; q < 2; q++) {
    int n = (wave * 2 + q) * 16 + l;
    v8f acc = bcast8(b2[n]);
    for (int kk = 0; kk < 64; kk++) {
      int kb = kk * 4 + half * 2;
      v2f a;  a.x = h1b[l * 256 + kb]; a.y = h1b[l * 256 + kb + 1];
      v2f bb; bb.x = W2[(size_t)kb * 256 + n]; bb.y = W2[(size_t)(kb + 1) * 256 + n];
      acc = __builtin_amdgcn_wmma_f32_16x16x4_f32(false, a, false, bb, (short)0, acc,
                                                  false, false);
    }
    for (int r = 0; r < 8; r++) h2b[(r + half * 8) * 256 + n] = fmaxf(acc[r], 0.f);
  }
  __syncthreads();

  // Layer 3: [16,256] @ [256,48(47)]
  if (wave < 3) {
    int n = wave * 16 + l;
    v8f acc = bcast8((n < FOUT) ? b3[n] : 0.f);
    for (int kk = 0; kk < 64; kk++) {
      int kb = kk * 4 + half * 2;
      v2f a;  a.x = h2b[l * 256 + kb]; a.y = h2b[l * 256 + kb + 1];
      v2f bb;
      bb.x = (n < FOUT) ? W3[(size_t)kb * FOUT + n] : 0.f;
      bb.y = (n < FOUT) ? W3[(size_t)(kb + 1) * FOUT + n] : 0.f;
      acc = __builtin_amdgcn_wmma_f32_16x16x4_f32(false, a, false, bb, (short)0, acc,
                                                  false, false);
    }
    for (int r = 0; r < 8; r++) ob[(r + half * 8) * 48 + n] = acc[r];
  }
  __syncthreads();

  // Head: expected offsets via softmax over 21 bins; clip; write outputs.
  if (tid < 16) {
    int j = tid, a = abase + j;
    const float* o = ob + j * 48;
    float st0 = anch[a * 2], en0 = anch[a * 2 + 1];
    float mx = -1e30f;
    for (int k = 0; k < DBINS; k++) mx = fmaxf(mx, o[k]);
    float se = 0.f, dot = 0.f;
    for (int k = 0; k < DBINS; k++) {
      float e = expf(o[k] - mx); se += e; dot += e * startw[s * DBINS + k];
    }
    float so = dot / se;
    mx = -1e30f;
    for (int k = 0; k < DBINS; k++) mx = fmaxf(mx, o[DBINS + k]);
    se = 0.f; dot = 0.f;
    for (int k = 0; k < DBINS; k++) {
      float e = expf(o[DBINS + k] - mx); se += e; dot += e * endw[s * DBINS + k];
    }
    float eo = dot / se;
    outp[a * 2 + 0] = fminf(fmaxf(st0 + so, 0.f), al);
    outp[a * 2 + 1] = fminf(fmaxf(en0 + eo, 0.f), al);
    outp[2 * NA + a] = o[2 * DBINS];
    for (int c = 0; c < 4; c++) outp[3 * NA + a * 4 + c] = o[2 * DBINS + 1 + c];
  }
}

// ---------------------------------------------------------------------------
extern "C" void kernel_launch(void* const* d_in, const int* in_sizes, int n_in,
                              void* d_out, int out_size, void* d_ws, size_t ws_size,
                              hipStream_t stream) {
  (void)in_sizes; (void)n_in; (void)out_size; (void)ws_size;
  const float* emb    = (const float*)d_in[0];
  const float* tp     = (const float*)d_in[1];
  const float* npred  = (const float*)d_in[2];
  const float* audio  = (const float*)d_in[3];
  const float* anchv  = (const float*)d_in[4];
  const float* kern   = (const float*)d_in[5];
  const float* fWih   = (const float*)d_in[6];
  const float* fWhh   = (const float*)d_in[7];
  const float* fbih   = (const float*)d_in[8];
  const float* fbhh   = (const float*)d_in[9];
  const float* aWih   = (const float*)d_in[10];
  const float* aWhh   = (const float*)d_in[11];
  const float* abih   = (const float*)d_in[12];
  const float* abhh   = (const float*)d_in[13];
  const float* startw = (const float*)d_in[14];
  const float* endw   = (const float*)d_in[15];
  const float* W1     = (const float*)d_in[16];
  const float* b1     = (const float*)d_in[17];
  const float* W2     = (const float*)d_in[18];
  const float* b2     = (const float*)d_in[19];
  const float* W3     = (const float*)d_in[20];
  const float* b3     = (const float*)d_in[21];

  float* ws    = (float*)d_ws;
  float* x     = ws;                               // 512*128
  float* GI    = x + N_NODES * DF;                 // 6*512*384
  float* abn   = GI + (size_t)6 * N_NODES * G3;    // 512
  float* tposb = abn + N_NODES;                    // 512
  float* hfeat = tposb + N_NODES;                  // 192*256
  float* habn  = hfeat + NA * 2 * HF;              // 384
  int* order   = (int*)(habn + NA * 2);            // 192*512
  int* lens    = order + NA * N_NODES;             // 192
  __bf16* whhbf = (__bf16*)(lens + NA);            // 6*384*128 bf16

  k_prep<<<N_NODES, DF, 0, stream>>>(emb, tp, npred, kern, audio, x, abn, tposb);
  k_order<<<NA, 32, 0, stream>>>(tposb, anchv, order, lens);
  k_gi<<<dim3(32, 144), 32, 0, stream>>>(x, fWih, fbih, GI);
  k_cvt<<<(6 * G3 * DF + 255) / 256, 256, 0, stream>>>(fWhh, whhbf);
  size_t smem = (size_t)G3 * HF * 2 + 16 * HF * 4 + 16 * G3 * 4;  // 131072 B
  k_gru_feat<<<24, 256, smem, stream>>>(GI, whhbf, fbhh, order, lens, hfeat);
  k_gru_abn<<<1, 384, 0, stream>>>(abn, aWih, aWhh, abih, abhh, order, lens, habn);
  k_mlp<<<12, 256, 0, stream>>>(hfeat, habn, anchv, audio, W1, b1, W2, b2, W3, b3,
                                startw, endw, (float*)d_out);
}
